// VQVAE_83683142795650
// MI455X (gfx1250) — matrix operand order repaired and checked by reading, not compiled, not run on hardware
//
#include <hip/hip_runtime.h>

typedef _Float16 v16h __attribute__((ext_vector_type(16)));
typedef float    v8f  __attribute__((ext_vector_type(8)));

// ---------------------------------------------------------------------------
// WMMA f16 fragment index maps (CDNA5 ISA 7.12.2, wave32)
//  A 16x32 (MxK): lane<16 -> M=lane, VGPR j holds K pair:
//      pairBase = (j<4 ? 2j : 16+2(j-4)) + 8*(lane>=16)
//  B 32x16 (KxN): lane&15 -> N, VGPR j holds K pair: same pairBase as A
//  C/D 16x16:     lane&15 -> N, VGPR v -> M = v + 8*(lane>=16)
// ---------------------------------------------------------------------------
__device__ __forceinline__ int wmma_ka(int e, int hi) {
  int j = e >> 1;
  int pb = ((j < 4) ? (2 * j) : (16 + 2 * (j - 4))) + (hi << 3);
  return pb + (e & 1);
}
__device__ __forceinline__ int wmma_kb(int e, int hi) {
  int j = e >> 1;
  return ((j < 4) ? (2 * j) : (16 + 2 * (j - 4))) + (hi << 3) + (e & 1);
}

union FragU { uint4 q[2]; v16h v; };

// ---------------------------------------------------------------------------
// Implicit-GEMM convolution (NCHW), fused bias + ReLU + optional residual.
// Block = 8 waves sharing one (image, 32-channel Cout tile), 512 pixels.
// Per K-step the 32x32 f32 weight tile is staged once into LDS as f16,
// pre-swizzled into A-fragment order; each lane reads its fragment as two
// 16B LDS vectors. B (im2col activations) loaded per-lane from global.
// R/STRIDE/PAD compile-time: (1,1,0), (3,1,1), (4,2,1).
// ---------------------------------------------------------------------------
template <int R, int STRIDE, int PAD>
__global__ __launch_bounds__(256) void wmma_conv_kernel(
    const float* __restrict__ in, const float* __restrict__ wgt,
    const float* __restrict__ bias, const float* __restrict__ res,
    float* __restrict__ out,
    int Nimg, int Cin, int Cout, int Hin, int Win, int Hout, int Wout,
    int relu)
{
  __shared__ uint4 smA4[2][32][2];                 // 2 frags x 32 lanes x 32B
  _Float16* smAh = (_Float16*)smA4;

  const int tid  = threadIdx.x;
  const int lane = tid & 31;
  const int wid  = tid >> 5;

  const int HWo       = Hout * Wout;
  const int pixBlocks = HWo >> 9;                  // 512 pixels per block
  const int coTiles   = Cout >> 5;
  const int blocksPerImg = pixBlocks * coTiles;

  const int n = blockIdx.x / blocksPerImg;
  int rem = blockIdx.x - n * blocksPerImg;
  const int ct  = rem / pixBlocks;
  const int co0 = ct << 5;
  const int p0  = ((rem - ct * pixBlocks) << 9) + (wid << 6);

  constexpr int RR = R * R;
  const int K    = Cin * RR;
  const int row  = lane & 15;
  const int hi   = lane >> 4;
  const int HWin = Hin * Win;
  const float* inN = in + (size_t)n * Cin * HWin;

  // per-fragment pixel coordinates for this lane's column
  int pix[4], iy0[4], ix0[4];
#pragma unroll
  for (int nf = 0; nf < 4; ++nf) {
    int p = p0 + (nf << 4) + row;
    pix[nf] = p;
    int oy = p / Wout;
    int ox = p - oy * Wout;
    iy0[nf] = oy * STRIDE - PAD;
    ix0[nf] = ox * STRIDE - PAD;
  }

  v8f acc[2][4];
  const v8f vzero = {0.f, 0.f, 0.f, 0.f, 0.f, 0.f, 0.f, 0.f};
#pragma unroll
  for (int mf = 0; mf < 2; ++mf)
#pragma unroll
    for (int nf = 0; nf < 4; ++nf)
      acc[mf][nf] = vzero;

  const int Kfull = K & ~31;

  // ---------------- main K loop: every k in range, no guards ----------------
  for (int kb = 0; kb < Kfull; kb += 32) {
    __syncthreads();                               // previous tile consumed
    // ---- cooperative weight staging: 1024 f32 -> swizzled f16 in LDS ----
#pragma unroll
    for (int t = 0; t < 4; ++t) {
      int ei = (t << 8) + tid;                     // 0..1023, c fastest
      int r  = ei >> 5;
      int c  = ei & 31;
      float w = wgt[(co0 + r) * K + kb + c];
      if (t == 0 && kb + 32 < K)                   // global_prefetch_b8
        __builtin_prefetch(&wgt[(co0 + r) * K + kb + 32 + c], 0, 0);
      int mf  = r >> 4;
      int lr  = r & 15;
      int h2  = (c >> 3) & 1;
      int ln  = lr + (h2 << 4);
      int cc  = c - (h2 << 3);
      int j   = (cc < 8) ? (cc >> 1) : (4 + ((cc - 16) >> 1));
      smAh[(((mf << 5) + ln) << 4) + (j << 1) + (c & 1)] = (_Float16)w;
    }
    __syncthreads();

    FragU fa0, fa1;
    fa0.q[0] = smA4[0][lane][0]; fa0.q[1] = smA4[0][lane][1];
    fa1.q[0] = smA4[1][lane][0]; fa1.q[1] = smA4[1][lane][1];

    v16h b[4];
#pragma unroll
    for (int e = 0; e < 16; ++e) {
      int kbi = kb + wmma_kb(e, hi);
      if (R == 1) {
#pragma unroll
        for (int nf = 0; nf < 4; ++nf)
          b[nf][e] = (_Float16)inN[kbi * HWin + pix[nf]];
      } else {
        int ci = kbi / RR;                         // compile-time RR
        int rs = kbi - ci * RR;
        int r  = rs / R;
        int s  = rs - r * R;
        const float* inC = inN + ci * HWin;
#pragma unroll
        for (int nf = 0; nf < 4; ++nf) {
          int iy = iy0[nf] + r;
          int ix = ix0[nf] + s;
          bool ok = ((unsigned)iy < (unsigned)Hin) & ((unsigned)ix < (unsigned)Win);
          b[nf][e] = (_Float16)(ok ? inC[iy * Win + ix] : 0.f);
        }
      }
    }
#pragma unroll
    for (int nf = 0; nf < 4; ++nf) {
      acc[0][nf] = __builtin_amdgcn_wmma_f32_16x16x32_f16(
          false, fa0.v, false, b[nf], (short)0, acc[0][nf], false, false);
      acc[1][nf] = __builtin_amdgcn_wmma_f32_16x16x32_f16(
          false, fa1.v, false, b[nf], (short)0, acc[1][nf], false, false);
    }
  }

  // ---------------- guarded tail (only enc.l1: K=48) ----------------
  if (Kfull < K) {
    const int kb = Kfull;
    __syncthreads();
#pragma unroll
    for (int t = 0; t < 4; ++t) {
      int ei = (t << 8) + tid;
      int r  = ei >> 5;
      int c  = ei & 31;
      float w = (kb + c < K) ? wgt[(co0 + r) * K + kb + c] : 0.f;
      int mf  = r >> 4;
      int lr  = r & 15;
      int h2  = (c >> 3) & 1;
      int ln  = lr + (h2 << 4);
      int cc  = c - (h2 << 3);
      int j   = (cc < 8) ? (cc >> 1) : (4 + ((cc - 16) >> 1));
      smAh[(((mf << 5) + ln) << 4) + (j << 1) + (c & 1)] = (_Float16)w;
    }
    __syncthreads();

    FragU fa0, fa1;
    fa0.q[0] = smA4[0][lane][0]; fa0.q[1] = smA4[0][lane][1];
    fa1.q[0] = smA4[1][lane][0]; fa1.q[1] = smA4[1][lane][1];

    v16h b[4];
#pragma unroll
    for (int e = 0; e < 16; ++e) {
      int kbi = kb + wmma_kb(e, hi);
      bool kbv = (kbi < K);
      int kc = kbv ? kbi : 0;
      int ci = kc / RR;
      int rs = kc - ci * RR;
      int r  = rs / R;
      int s  = rs - r * R;
      const float* inC = inN + ci * HWin;
#pragma unroll
      for (int nf = 0; nf < 4; ++nf) {
        int iy = iy0[nf] + r;
        int ix = ix0[nf] + s;
        bool ok = kbv & ((unsigned)iy < (unsigned)Hin) & ((unsigned)ix < (unsigned)Win);
        b[nf][e] = (_Float16)(ok ? inC[iy * Win + ix] : 0.f);
      }
    }
#pragma unroll
    for (int nf = 0; nf < 4; ++nf) {
      acc[0][nf] = __builtin_amdgcn_wmma_f32_16x16x32_f16(
          false, fa0.v, false, b[nf], (short)0, acc[0][nf], false, false);
      acc[1][nf] = __builtin_amdgcn_wmma_f32_16x16x32_f16(
          false, fa1.v, false, b[nf], (short)0, acc[1][nf], false, false);
    }
  }

  // ---- epilogue: bias + ReLU (+ residual) ----
#pragma unroll
  for (int mf = 0; mf < 2; ++mf) {
#pragma unroll
    for (int nf = 0; nf < 4; ++nf) {
#pragma unroll
      for (int v = 0; v < 8; ++v) {
        int m = co0 + (mf << 4) + v + (hi << 3);
        float val = acc[mf][nf][v] + bias[m];
        if (relu) val = fmaxf(val, 0.0f);
        int oidx = (n * Cout + m) * HWo + pix[nf];
        if (res) val += res[oidx];
        out[oidx] = val;
      }
    }
  }
}

// ---------------------------------------------------------------------------
// Codebook: ||E_k||^2 precompute
// ---------------------------------------------------------------------------
__global__ void vq_norms_kernel(const float* __restrict__ embed,
                                float* __restrict__ norms) {
  int t = blockIdx.x * blockDim.x + threadIdx.x;
  if (t >= 512) return;
  float s = 0.f;
  for (int k = 0; k < 64; ++k) {
    float v = embed[t * 64 + k];
    s += v * v;
  }
  norms[t] = s;
}

// ---------------------------------------------------------------------------
// Codebook argmin via WMMA: each wave owns 16 pixels; loops 32 code tiles of
// 16 codes, K=64 in two 32-steps; score = ||E||^2 - 2 z.E; cross-half merge
// with shfl_xor(16) (wave32).
// ---------------------------------------------------------------------------
__global__ __launch_bounds__(256) void vq_argmin_kernel(
    const float* __restrict__ z, const float* __restrict__ embed,
    const float* __restrict__ norms, int* __restrict__ idx,
    int Npix, int HWpix)
{
  const int lane = threadIdx.x & 31;
  const int wave = blockIdx.x * (blockDim.x >> 5) + (threadIdx.x >> 5);
  const int P0 = wave << 4;
  if (P0 >= Npix) return;
  const int n  = P0 / HWpix;
  const int p0 = P0 - n * HWpix;
  const int row = lane & 15;
  const int hi  = lane >> 4;
  const float* zimg = z + (size_t)n * 64 * HWpix;

  // cache both K-step B fragments for this wave's 16 pixels
  v16h bk0, bk1;
#pragma unroll
  for (int e = 0; e < 16; ++e) {
    int k = wmma_kb(e, hi);
    bk0[e] = (_Float16)zimg[k * HWpix + p0 + row];
    bk1[e] = (_Float16)zimg[(k + 32) * HWpix + p0 + row];
  }

  float bestVal = 3.4e38f;
  int bestIdx = 0;
  const v8f vzero = {0.f, 0.f, 0.f, 0.f, 0.f, 0.f, 0.f, 0.f};
  for (int t = 0; t < 512; t += 16) {
    v16h a0, a1;
#pragma unroll
    for (int e = 0; e < 16; ++e) {
      int k = wmma_ka(e, hi);
      a0[e] = (_Float16)embed[(t + row) * 64 + k];
      a1[e] = (_Float16)embed[(t + row) * 64 + k + 32];
    }
    v8f acc = vzero;
    acc = __builtin_amdgcn_wmma_f32_16x16x32_f16(false, a0, false, bk0, (short)0, acc, false, false);
    acc = __builtin_amdgcn_wmma_f32_16x16x32_f16(false, a1, false, bk1, (short)0, acc, false, false);
#pragma unroll
    for (int v = 0; v < 8; ++v) {
      int m = t + v + (hi << 3);
      float sc = norms[m] - 2.0f * acc[v];
      if (sc < bestVal) { bestVal = sc; bestIdx = m; }
    }
  }
  float ov = __shfl_xor(bestVal, 16, 32);
  int   oi = __shfl_xor(bestIdx, 16, 32);
  if (ov < bestVal || (ov == bestVal && oi < bestIdx)) { bestVal = ov; bestIdx = oi; }
  if (lane < 16) idx[P0 + row] = bestIdx;
}

// ---------------------------------------------------------------------------
// Straight-through (eval): rep[n,c,p] = E[idx[n,p], c]   (NCHW output)
// ---------------------------------------------------------------------------
__global__ void vq_gather_kernel(const int* __restrict__ idx,
                                 const float* __restrict__ embed,
                                 float* __restrict__ rep,
                                 int total, int HWpix) {
  int i = blockIdx.x * blockDim.x + threadIdx.x;
  if (i >= total) return;
  int p = i % HWpix;
  int c = (i / HWpix) & 63;            // LAT = 64
  int n = i / (HWpix * 64);
  rep[i] = embed[idx[n * HWpix + p] * 64 + c];
}

// ---------------------------------------------------------------------------
// ConvTranspose2d(k=4, s=2, p=1) + ReLU, direct (2x2 valid taps per output).
// ---------------------------------------------------------------------------
__global__ void convt_relu_kernel(const float* __restrict__ in,
                                  const float* __restrict__ wgt,
                                  const float* __restrict__ bias,
                                  float* __restrict__ out,
                                  int Nimg, int Cin, int Cout, int Hin, int Win) {
  const int Hout = Hin * 2, Wout = Win * 2;
  const int total = Nimg * Cout * Hout * Wout;
  int i = blockIdx.x * blockDim.x + threadIdx.x;
  if (i >= total) return;
  int ox = i % Wout;
  int t1 = i / Wout;
  int oy = t1 % Hout;
  int t2 = t1 / Hout;
  int co = t2 % Cout;
  int n  = t2 / Cout;

  const int HWin = Hin * Win;
  const float* inN = in + (size_t)n * Cin * HWin;
  const float* wco = wgt + (size_t)co * Cin * 16;
  float acc = bias[co];
#pragma unroll
  for (int r = 0; r < 4; ++r) {
    int ty = oy + r - 2;
    if (ty < 0 || (ty & 1)) continue;
    int iy = ty >> 1;
    if (iy >= Hin) continue;
#pragma unroll
    for (int s = 0; s < 4; ++s) {
      int tx = ox + s - 2;
      if (tx < 0 || (tx & 1)) continue;
      int ix = tx >> 1;
      if (ix >= Win) continue;
      const float* ip = inN + iy * Win + ix;
      const float* wp = wco + r * 4 + s;
      float a = 0.f;
      for (int ci = 0; ci < Cin; ++ci)
        a += wp[ci * 16] * ip[(size_t)ci * HWin];
      acc += a;
    }
  }
  out[i] = fmaxf(acc, 0.0f);
}

// ---------------------------------------------------------------------------
// Host side: full VQ-VAE forward pipeline.
// Input flattening assumes setup_inputs() dict insertion order:
//   0: x | enc: 1-4 l1/l2 (w,b) | 5-16 l3 blocks | 17-28 l4 blocks
//   cb: 29 proj.w 30 proj.b 31 embed
//   dec: 32-43 l1 blocks | 44-45 l2 | 46-57 l3 blocks | 58-59 l4
// ---------------------------------------------------------------------------
extern "C" void kernel_launch(void* const* d_in, const int* in_sizes, int n_in,
                              void* d_out, int out_size, void* d_ws, size_t ws_size,
                              hipStream_t stream) {
  (void)in_sizes; (void)n_in; (void)out_size; (void)ws_size;
  auto F = [&](int i) { return (const float*)d_in[i]; };

  // Workspace layout (fp32 activations, ping-pong reuse)
  char* ws = (char*)d_ws;
  size_t off = 0;
  auto alloc = [&](size_t bytes) { float* p = (float*)(ws + off); off += (bytes + 255) & ~(size_t)255; return p; };
  float* A0 = alloc((size_t)32 * 64 * 128 * 128 * 4);   // 134 MB
  float* A1 = alloc((size_t)32 * 64 * 128 * 128 * 4);   // 134 MB
  float* C0 = alloc((size_t)32 * 128 * 64 * 64 * 4);    // 67 MB (aliased as 32ch@128x128 later)
  float* C1 = alloc((size_t)32 * 128 * 64 * 64 * 4);    // 67 MB
  float* D0 = alloc((size_t)32 * 64 * 64 * 64 * 4);     // 33.5 MB
  float* D1 = alloc((size_t)32 * 64 * 64 * 64 * 4);     // 33.5 MB
  float* NORMS = alloc(512 * 4);
  int*   IDX   = (int*)alloc((size_t)32 * 64 * 64 * 4);
  float* E0 = C0;  // 32*32*128*128 fp32 == C0 size
  float* E1 = C1;

  auto conv = [&](const float* in, const float* w, const float* b, const float* res,
                  float* out, int Cin, int Cout, int Hin, int Win,
                  int Hout, int Wout, int R, int relu) {
    int blocks = 32 * (Cout / 32) * ((Hout * Wout) / 512);  // 8 waves/block
    if (R == 1)
      wmma_conv_kernel<1, 1, 0><<<blocks, 256, 0, stream>>>(
          in, w, b, res, out, 32, Cin, Cout, Hin, Win, Hout, Wout, relu);
    else if (R == 3)
      wmma_conv_kernel<3, 1, 1><<<blocks, 256, 0, stream>>>(
          in, w, b, res, out, 32, Cin, Cout, Hin, Win, Hout, Wout, relu);
    else
      wmma_conv_kernel<4, 2, 1><<<blocks, 256, 0, stream>>>(
          in, w, b, res, out, 32, Cin, Cout, Hin, Win, Hout, Wout, relu);
  };

  const float* x = F(0);

  // ---------------- encoder ----------------
  conv(x,  F(1), F(2), nullptr, A0,  3,  64, 256, 256, 128, 128, 4, 1); // enc.l1
  conv(A0, F(3), F(4), nullptr, C0, 64, 128, 128, 128,  64,  64, 4, 1); // enc.l2
  // enc.l3 block0
  conv(C0, F(5),  F(6),  nullptr, D0, 128, 64, 64, 64, 64, 64, 1, 1);
  conv(D0, F(7),  F(8),  nullptr, D1,  64, 64, 64, 64, 64, 64, 3, 1);
  conv(D1, F(9),  F(10), C0,      C1,  64,128, 64, 64, 64, 64, 1, 1);
  // enc.l3 block1
  conv(C1, F(11), F(12), nullptr, D0, 128, 64, 64, 64, 64, 64, 1, 1);
  conv(D0, F(13), F(14), nullptr, D1,  64, 64, 64, 64, 64, 64, 3, 1);
  conv(D1, F(15), F(16), C1,      C0,  64,128, 64, 64, 64, 64, 1, 1);
  // enc.l4 block0
  conv(C0, F(17), F(18), nullptr, D0, 128, 64, 64, 64, 64, 64, 1, 1);
  conv(D0, F(19), F(20), nullptr, D1,  64, 64, 64, 64, 64, 64, 3, 1);
  conv(D1, F(21), F(22), C0,      C1,  64,128, 64, 64, 64, 64, 1, 1);
  // enc.l4 block1
  conv(C1, F(23), F(24), nullptr, D0, 128, 64, 64, 64, 64, 64, 1, 1);
  conv(D0, F(25), F(26), nullptr, D1,  64, 64, 64, 64, 64, 64, 3, 1);
  conv(D1, F(27), F(28), C1,      C0,  64,128, 64, 64, 64, 64, 1, 1);   // z_e = C0

  // ---------------- codebook ----------------
  const float* embed = F(31);
  conv(C0, F(29), F(30), nullptr, D0, 128, 64, 64, 64, 64, 64, 1, 0);   // z = proj(z_e), no relu
  vq_norms_kernel<<<2, 256, 0, stream>>>(embed, NORMS);
  {
    int Npix = 32 * 64 * 64;                    // 131072
    int blocks = (Npix / 16) / 8;               // 1024
    vq_argmin_kernel<<<blocks, 256, 0, stream>>>(D0, embed, NORMS, IDX, Npix, 64 * 64);
  }
  {
    int total = 32 * 64 * 64 * 64;              // 8.39M
    vq_gather_kernel<<<total / 256, 256, 0, stream>>>(IDX, embed, D1, total, 64 * 64);  // rep = D1
  }

  // ---------------- decoder ----------------
  // dec.l1 block0 (no shortcut)
  conv(D1, F(32), F(33), nullptr, D0, 64, 64, 64, 64, 64, 64, 1, 1);
  conv(D0, F(34), F(35), nullptr, D1, 64, 64, 64, 64, 64, 64, 3, 1);
  conv(D1, F(36), F(37), nullptr, C0, 64,128, 64, 64, 64, 64, 1, 1);
  // dec.l1 block1 (shortcut)
  conv(C0, F(38), F(39), nullptr, D0,128, 64, 64, 64, 64, 64, 1, 1);
  conv(D0, F(40), F(41), nullptr, D1, 64, 64, 64, 64, 64, 64, 3, 1);
  conv(D1, F(42), F(43), C0,      C1, 64,128, 64, 64, 64, 64, 1, 1);
  // dec.l2: ConvT 128->64, 64x64 -> 128x128, relu
  {
    int total = 32 * 64 * 128 * 128;
    convt_relu_kernel<<<total / 256, 256, 0, stream>>>(C1, F(44), F(45), A0, 32, 128, 64, 64, 64);
  }
  // dec.l3 block0 (inter=32)
  conv(A0, F(46), F(47), nullptr, E0, 64, 32, 128, 128, 128, 128, 1, 1);
  conv(E0, F(48), F(49), nullptr, E1, 32, 32, 128, 128, 128, 128, 3, 1);
  conv(E1, F(50), F(51), A0,      A1, 32, 64, 128, 128, 128, 128, 1, 1);
  // dec.l3 block1
  conv(A1, F(52), F(53), nullptr, E0, 64, 32, 128, 128, 128, 128, 1, 1);
  conv(E0, F(54), F(55), nullptr, E1, 32, 32, 128, 128, 128, 128, 3, 1);
  conv(E1, F(56), F(57), A1,      A0, 32, 64, 128, 128, 128, 128, 1, 1);
  // dec.l4: ConvT 64->3, 128x128 -> 256x256, relu -> d_out
  {
    int total = 32 * 3 * 256 * 256;
    convt_relu_kernel<<<total / 256, 256, 0, stream>>>(A0, F(58), F(59), (float*)d_out,
                                                       32, 64, 3, 128, 128);
  }
}
